// MetaCAArchOpt_77360950935628
// MI455X (gfx1250) — compile-verified, matrix operand order checked by reading, not run on hardware
//
#include <hip/hip_runtime.h>
#include <hip/hip_bf16.h>
#include <math.h>

typedef __attribute__((ext_vector_type(16))) _Float16 v16h;
typedef __attribute__((ext_vector_type(8)))  float    v8f;

#define D_MODEL 64
#define HID     128      // 2*D
#define K1      192      // 3*D
#define N_RULES 4
#define C_DIM   128
#define BB      32
#define TT      8192
#define LN_EPS  1e-5f

struct Ctrl {
    float rw[N_RULES];
    float alpha;
    int   n_evolve;
    float pad[2];
};

// ws layout (bytes):
//   0      : Ctrl
//   64     : W1h  f16 frags: 4 rules * 8 nblk * 6 kchunk * 512 = 98304 f16 (196608 B)
//   196672 : W2h  f16 frags: 4 rules * 4 nblk * 4 kchunk * 512 = 32768 f16 ( 65536 B)
//   524288 : bufA floats: 32*8192*64 (64 MB)
#define WS_W1H_OFF   64
#define WS_W2H_OFF   (64 + 196608)
#define WS_BUFA_OFF  (1 << 19)

// ---------------- branchless transcendentals (no EXEC divergence) ----------
__device__ __forceinline__ float fast_exp2(float x) { return __builtin_amdgcn_exp2f(x); }
__device__ __forceinline__ float fast_rcp(float x)  { return __builtin_amdgcn_rcpf(x); }

// tanh(x) = 1 - 2/(exp(2x)+1); saturates correctly at +/-inf of the exp
__device__ __forceinline__ float tanh_fast(float x) {
    float e = fast_exp2(x * 2.8853900817779268f);   // 2*log2(e)
    return 1.0f - 2.0f * fast_rcp(e + 1.0f);
}

// GELU exact form via Abramowitz-Stegun 7.1.26 erf (|err| < 1.5e-7), branchless
__device__ __forceinline__ float gelu_fast(float x) {
    float z = fabsf(x) * 0.70710678118654752440f;   // |x|/sqrt(2)
    float t = fast_rcp(__builtin_fmaf(0.3275911f, z, 1.0f));
    float p = __builtin_fmaf(t, 1.061405429f, -1.453152027f);
    p = __builtin_fmaf(t, p,  1.421413741f);
    p = __builtin_fmaf(t, p, -0.284496736f);
    p = __builtin_fmaf(t, p,  0.254829592f);
    p = p * t;
    float g = fast_exp2(-z * z * 1.4426950408889634f);  // exp(-z^2)
    float erfz = __builtin_copysignf(1.0f - p * g, x);  // erf(x/sqrt(2))
    return 0.5f * x * (1.0f + erfz);
}

// ---------------------------------------------------------------------------
// Control kernel: derive rule_weights, n_evolve, alpha from c_state (tiny).
// ---------------------------------------------------------------------------
__global__ __launch_bounds__(64) void ctrl_kernel(
    const float* __restrict__ c,
    const float* __restrict__ sel_w1, const float* __restrict__ sel_b1,
    const float* __restrict__ sel_w2, const float* __restrict__ sel_b2,
    const float* __restrict__ st_w1,  const float* __restrict__ st_b1,
    const float* __restrict__ st_w2,  const float* __restrict__ st_b2,
    const float* __restrict__ res_w1, const float* __restrict__ res_b1,
    const float* __restrict__ res_w2, const float* __restrict__ res_b2,
    Ctrl* __restrict__ ctrl)
{
    __shared__ float cs[C_DIM];
    __shared__ float hid[64];
    const int tid = threadIdx.x;

    cs[tid]      = c[tid];
    cs[tid + 64] = c[tid + 64];
    __syncthreads();

    // selector: 128 -> 64 (GELU) -> 4, softmax
    {
        float s = sel_b1[tid];
        for (int k = 0; k < C_DIM; ++k) s += cs[k] * sel_w1[k * 64 + tid];
        hid[tid] = gelu_fast(s);
    }
    __syncthreads();
    if (tid == 0) {
        float lg[N_RULES];
        float mx = -1e30f;
        for (int j = 0; j < N_RULES; ++j) {
            float s = sel_b2[j];
            for (int k = 0; k < 64; ++k) s += hid[k] * sel_w2[k * N_RULES + j];
            lg[j] = s; mx = fmaxf(mx, s);
        }
        float den = 0.f;
        for (int j = 0; j < N_RULES; ++j) { lg[j] = expf(lg[j] - mx); den += lg[j]; }
        for (int j = 0; j < N_RULES; ++j) ctrl->rw[j] = lg[j] / den;
    }
    __syncthreads();

    // steps: 128 -> 32 (GELU) -> 7, softmax, expected step, round, clamp [2,8]
    if (tid < 32) {
        float s = st_b1[tid];
        for (int k = 0; k < C_DIM; ++k) s += cs[k] * st_w1[k * 32 + tid];
        hid[tid] = gelu_fast(s);
    }
    __syncthreads();
    if (tid == 0) {
        const float steps[7] = {2.f, 3.f, 4.f, 5.f, 6.f, 7.f, 8.f};
        float lg[7];
        float mx = -1e30f;
        for (int j = 0; j < 7; ++j) {
            float s = st_b2[j];
            for (int k = 0; k < 32; ++k) s += hid[k] * st_w2[k * 7 + j];
            lg[j] = s; mx = fmaxf(mx, s);
        }
        float den = 0.f;
        for (int j = 0; j < 7; ++j) { lg[j] = expf(lg[j] - mx); den += lg[j]; }
        float nsoft = 0.f;
        for (int j = 0; j < 7; ++j) nsoft += (lg[j] / den) * steps[j];
        int n = (int)(nsoft + 0.5f);           // trunc(x+0.5) for positive x
        n = n < 2 ? 2 : (n > 8 ? 8 : n);
        ctrl->n_evolve = n;
    }
    __syncthreads();

    // residual: 128 -> 32 (GELU) -> 1, sigmoid, alpha = 0.1 + 0.8*sig
    if (tid < 32) {
        float s = res_b1[tid];
        for (int k = 0; k < C_DIM; ++k) s += cs[k] * res_w1[k * 32 + tid];
        hid[tid] = gelu_fast(s);
    }
    __syncthreads();
    if (tid == 0) {
        float s = res_b2[0];
        for (int k = 0; k < 32; ++k) s += hid[k] * res_w2[k];
        float sig = 1.f / (1.f + expf(-s));
        ctrl->alpha = 0.1f + 0.8f * sig;
    }
}

// ---------------------------------------------------------------------------
// Pre-shuffle fp32 weights into f16 WMMA B-fragment layout.
// B frag k order used here (same as A): n = lane&15, base_k = (lane>>4)*16, elem i -> k=base_k+i
// Fragment = 512 f16; each lane's 16 f16 contiguous -> one aligned 32B load.
// ---------------------------------------------------------------------------
__global__ __launch_bounds__(256) void prep_kernel(
    const float* __restrict__ w1, const float* __restrict__ w2,
    _Float16* __restrict__ w1h, _Float16* __restrict__ w2h)
{
    int gid = blockIdx.x * 256 + threadIdx.x;
    if (gid < 98304) {                           // W1: 4*8*6 frags
        int f = gid >> 9, rem = gid & 511;
        int lane = rem >> 4, i = rem & 15;
        int ck = f % 6, t = f / 6;
        int nb = t & 7, r = t >> 3;
        int n = nb * 16 + (lane & 15);
        int k = ck * 32 + ((lane >> 4) << 4) + i;
        w1h[gid] = (_Float16)w1[(r * K1 + k) * HID + n];
    } else if (gid < 98304 + 32768) {            // W2: 4*4*4 frags
        int g2 = gid - 98304;
        int f = g2 >> 9, rem = g2 & 511;
        int lane = rem >> 4, i = rem & 15;
        int ck = f & 3, t = f >> 2;
        int nb = t & 3, r = t >> 2;
        int n = nb * 16 + (lane & 15);
        int k = ck * 32 + ((lane >> 4) << 4) + i;
        w2h[g2] = (_Float16)w2[(r * HID + k) * D_MODEL + n];
    }
}

// ---------------------------------------------------------------------------
// One CA evolve step. 4 waves/block, each wave owns TWO M=16 tiles (32 tokens,
// block covers 128 tokens). Each B-fragment clause feeds two WMMAs.
//   layer1: [16x192]x[192x128] -> 96 WMMAs per rule for the tile pair
//   layer2: [16x128]x[128x64]  -> 32 WMMAs per rule for the tile pair
// ---------------------------------------------------------------------------
__global__ __launch_bounds__(128) void evolve_kernel(
    const float* __restrict__ src, float* __restrict__ dst,
    const _Float16* __restrict__ w1h, const _Float16* __restrict__ w2h,
    const float* __restrict__ b1, const float* __restrict__ b2,
    const Ctrl* __restrict__ ctrl, int step)
{
    if (step >= ctrl->n_evolve) return;          // uniform across grid

    __shared__ __align__(32) _Float16 tileh[130 * 64];       // 128 tokens + halo (f16)
    __shared__ __align__(32) _Float16 hbuf[4][2][16 * HID];  // per-wave GELU output

    const int bIdx = blockIdx.x >> 6;            // batch row (8192/128 = 64 chunks)
    const int t0wg = (blockIdx.x & 63) << 7;     // first of 128 tokens
    const int tid  = threadIdx.x;

    // cooperative tile load with wrap halo; pack 2 f16 per dword store
    for (int idx = tid; idx < 130 * 32; idx += 128) {
        int j = idx >> 5, cp = (idx & 31) * 2;
        int tg = t0wg - 1 + j;
        tg = (tg < 0) ? tg + TT : (tg >= TT ? tg - TT : tg);
        const float2 v = *(const float2*)&src[((size_t)bIdx * TT + tg) * D_MODEL + cp];
        union { _Float16 h[2]; unsigned u; } pk;
        pk.h[0] = (_Float16)v.x; pk.h[1] = (_Float16)v.y;
        *(unsigned*)&tileh[j * 64 + cp] = pk.u;
    }
    __syncthreads();

    const int wave = tid >> 5, lane = tid & 31;
    const int m  = lane & 15;                    // A-row / D-col index
    const int hs = lane >> 4;                    // half-select
    const int row0 = wave * 32 + 1;              // local row of this wave's token 0

    const float alpha = ctrl->alpha;
    float rw[N_RULES];
    #pragma unroll
    for (int r = 0; r < N_RULES; ++r) rw[r] = ctrl->rw[r];

    // (1-alpha)*sum_r rw[r]*tanh(out_r) at (M = j+hs*8, N = nb*16+m)
    v8f fin0[4], fin1[4];
    #pragma unroll
    for (int nb = 0; nb < 4; ++nb)
        #pragma unroll
        for (int j = 0; j < 8; ++j) { fin0[nb][j] = 0.f; fin1[nb][j] = 0.f; }

    _Float16* h0 = hbuf[wave][0];
    _Float16* h1 = hbuf[wave][1];

    for (int r = 0; r < N_RULES; ++r) {
        // rebuild A-fragments from the f16 LDS tile (single 32B load each)
        v16h af0[6], af1[6];
        #pragma unroll
        for (int ck = 0; ck < 6; ++ck) {
            int seg = ck >> 1;
            int dt = (seg == 0) ? 0 : (seg == 1 ? -1 : 1);
            int cb = (ck & 1) * 32 + hs * 16;
            af0[ck] = *(const v16h*)&tileh[(row0 + m + dt) * 64 + cb];
            af1[ck] = *(const v16h*)&tileh[(row0 + 16 + m + dt) * 64 + cb];
        }

        // ---- layer 1: h = GELU(X @ W1_r + b1_r) ----
        #pragma unroll
        for (int nb = 0; nb < 8; ++nb) {
            const float bias = b1[r * HID + nb * 16 + m];
            v8f a0, a1;
            #pragma unroll
            for (int j = 0; j < 8; ++j) { a0[j] = bias; a1[j] = bias; }
            const _Float16* bb = w1h + (size_t)((r * 8 + nb) * 6) * 512 + lane * 16;
            v16h bf[6];
            #pragma unroll
            for (int ck = 0; ck < 6; ++ck) bf[ck] = *(const v16h*)(bb + ck * 512);
            #pragma unroll
            for (int ck = 0; ck < 6; ++ck)
                a0 = __builtin_amdgcn_wmma_f32_16x16x32_f16(
                        false, af0[ck], false, bf[ck], (short)0, a0, false, false);
            #pragma unroll
            for (int ck = 0; ck < 6; ++ck)
                a1 = __builtin_amdgcn_wmma_f32_16x16x32_f16(
                        false, af1[ck], false, bf[ck], (short)0, a1, false, false);
            const int n = nb * 16 + m;
            #pragma unroll
            for (int j = 0; j < 8; ++j) {
                const int M = j + hs * 8;
                h0[M * HID + n] = (_Float16)gelu_fast(a0[j]);
                h1[M * HID + n] = (_Float16)gelu_fast(a1[j]);
            }
        }
        __syncthreads();   // make h visible for the re-layout read

        // ---- re-layout h as A-fragments for layer 2 ----
        v16h a20[4], a21[4];
        #pragma unroll
        for (int ck = 0; ck < 4; ++ck) {
            a20[ck] = *(const v16h*)&h0[m * HID + ck * 32 + hs * 16];
            a21[ck] = *(const v16h*)&h1[m * HID + ck * 32 + hs * 16];
        }

        // ---- layer 2: out = tanh(h @ W2_r + b2_r), rule-weighted accumulate ----
        const float scale = (1.0f - alpha) * rw[r];
        #pragma unroll
        for (int nb = 0; nb < 4; ++nb) {
            const float bias = b2[r * D_MODEL + nb * 16 + m];
            v8f o0, o1;
            #pragma unroll
            for (int j = 0; j < 8; ++j) { o0[j] = bias; o1[j] = bias; }
            const _Float16* bb = w2h + (size_t)((r * 4 + nb) * 4) * 512 + lane * 16;
            v16h bf[4];
            #pragma unroll
            for (int ck = 0; ck < 4; ++ck) bf[ck] = *(const v16h*)(bb + ck * 512);
            #pragma unroll
            for (int ck = 0; ck < 4; ++ck)
                o0 = __builtin_amdgcn_wmma_f32_16x16x32_f16(
                        false, a20[ck], false, bf[ck], (short)0, o0, false, false);
            #pragma unroll
            for (int ck = 0; ck < 4; ++ck)
                o1 = __builtin_amdgcn_wmma_f32_16x16x32_f16(
                        false, a21[ck], false, bf[ck], (short)0, o1, false, false);
            #pragma unroll
            for (int j = 0; j < 8; ++j) {
                fin0[nb][j] += scale * tanh_fast(o0[j]);
                fin1[nb][j] += scale * tanh_fast(o1[j]);
            }
        }
        __syncthreads();   // protect hbuf WAR before next rule overwrites it
    }

    // epilogue: cells' = alpha*cells_old(fp32 from global) + accumulated
    const size_t base = (size_t)bIdx * TT + t0wg + wave * 32;
    #pragma unroll
    for (int nb = 0; nb < 4; ++nb) {
        const int n = nb * 16 + m;
        #pragma unroll
        for (int j = 0; j < 8; ++j) {
            const int M = j + hs * 8;
            const size_t i0 = (base + M) * D_MODEL + n;
            const size_t i1 = (base + 16 + M) * D_MODEL + n;
            dst[i0] = __builtin_fmaf(alpha, src[i0], fin0[nb][j]);
            dst[i1] = __builtin_fmaf(alpha, src[i1], fin1[nb][j]);
        }
    }
}

// ---------------------------------------------------------------------------
// Final LayerNorm. One wave per 64-wide row, 2 elems per lane.
// Picks source buffer from n_evolve parity; always rewrites d_out fully.
// ---------------------------------------------------------------------------
__global__ __launch_bounds__(256) void ln_kernel(
    const float* __restrict__ bufA, const float* __restrict__ bufB,
    const float* __restrict__ g, const float* __restrict__ beta,
    const Ctrl* __restrict__ ctrl, float* __restrict__ out)
{
    const int n = ctrl->n_evolve;
    const float* src = (((n - 1) & 1) == 0) ? bufA : bufB;

    const int wave = threadIdx.x >> 5, lane = threadIdx.x & 31;
    const size_t row = (size_t)blockIdx.x * 8 + wave;
    const float* p = src + row * D_MODEL;

    float x0 = p[lane], x1 = p[lane + 32];
    float s = x0 + x1;
    #pragma unroll
    for (int o = 16; o > 0; o >>= 1) s += __shfl_xor(s, o, 32);
    float mu = s * (1.0f / 64.0f);
    float d0 = x0 - mu, d1 = x1 - mu;
    float v = d0 * d0 + d1 * d1;
    #pragma unroll
    for (int o = 16; o > 0; o >>= 1) v += __shfl_xor(v, o, 32);
    float inv = rsqrtf(v * (1.0f / 64.0f) + LN_EPS);

    out[row * D_MODEL + lane]      = d0 * inv * g[lane] + beta[lane];
    out[row * D_MODEL + lane + 32] = d1 * inv * g[lane + 32] + beta[lane + 32];
}

// ---------------------------------------------------------------------------
extern "C" void kernel_launch(void* const* d_in, const int* in_sizes, int n_in,
                              void* d_out, int out_size, void* d_ws, size_t ws_size,
                              hipStream_t stream) {
    (void)in_sizes; (void)n_in; (void)out_size; (void)ws_size;

    const float* cells   = (const float*)d_in[0];
    const float* c_state = (const float*)d_in[1];
    const float* rule_w1 = (const float*)d_in[2];
    const float* rule_b1 = (const float*)d_in[3];
    const float* rule_w2 = (const float*)d_in[4];
    const float* rule_b2 = (const float*)d_in[5];
    const float* sel_w1  = (const float*)d_in[6];
    const float* sel_b1  = (const float*)d_in[7];
    const float* sel_w2  = (const float*)d_in[8];
    const float* sel_b2  = (const float*)d_in[9];
    const float* st_w1   = (const float*)d_in[10];
    const float* st_b1   = (const float*)d_in[11];
    const float* st_w2   = (const float*)d_in[12];
    const float* st_b2   = (const float*)d_in[13];
    const float* res_w1  = (const float*)d_in[14];
    const float* res_b1  = (const float*)d_in[15];
    const float* res_w2  = (const float*)d_in[16];
    const float* res_b2  = (const float*)d_in[17];
    const float* ln_g    = (const float*)d_in[18];
    const float* ln_b    = (const float*)d_in[19];

    char* ws = (char*)d_ws;
    Ctrl*     ctrl = (Ctrl*)ws;
    _Float16* w1h  = (_Float16*)(ws + WS_W1H_OFF);
    _Float16* w2h  = (_Float16*)(ws + WS_W2H_OFF);
    float*    bufA = (float*)(ws + WS_BUFA_OFF);
    float*    bufB = (float*)d_out;

    ctrl_kernel<<<1, 64, 0, stream>>>(c_state,
        sel_w1, sel_b1, sel_w2, sel_b2,
        st_w1, st_b1, st_w2, st_b2,
        res_w1, res_b1, res_w2, res_b2, ctrl);

    prep_kernel<<<(98304 + 32768 + 255) / 256, 256, 0, stream>>>(rule_w1, rule_w2, w1h, w2h);

    // 8 launches; each uniformly no-ops when step >= n_evolve (2..8).
    for (int s = 0; s < 8; ++s) {
        const float* src = (s == 0) ? cells : ((s & 1) ? bufA : bufB);
        float*       dst = (s & 1) ? bufB : bufA;
        evolve_kernel<<<BB * (TT / 128), 128, 0, stream>>>(
            src, dst, w1h, w2h, rule_b1, rule_b2, ctrl, s);
    }

    ln_kernel<<<(BB * TT) / 8, 256, 0, stream>>>(bufA, bufB, ln_g, ln_b, ctrl, (float*)d_out);
}
